// RNN_13082470384316
// MI455X (gfx1250) — compile-verified, matrix operand order checked by reading, not run on hardware
//
#include <hip/hip_runtime.h>

typedef __bf16 bf16;
typedef bf16  v16bf __attribute__((ext_vector_type(16)));
typedef bf16  v8bf  __attribute__((ext_vector_type(8)));
typedef bf16  v4bf  __attribute__((ext_vector_type(4)));
typedef float v8f   __attribute__((ext_vector_type(8)));

#define T_STEPS   200
#define D_INPUT   300
#define D_HID     256
#define K_TOT     556          // D_INPUT + D_HID
#define K_PAD     576          // next multiple of 32
#define KT        18           // K_PAD / 32
#define KGRP      3            // K-tiles per load group (forces distinct A-frag registers)
#define M_BLK     16           // batch rows per block
#define A_STRIDE  584          // bf16 elements per A row (292 DWORDs -> conflict-free ds_load_b128)
#define NWAVES    16           // one wave per 16-col N tile (16*16 = 256 = D_HID)
#define NTHREADS  512
#define NCHUNK    1200         // 16 rows * 75 float4 per step

__device__ __forceinline__ float fast_tanh(float x) {
#if __has_builtin(__builtin_amdgcn_tanhf)
    return __builtin_amdgcn_tanhf(x);
#elif __has_builtin(__builtin_amdgcn_tanh_f32)
    return __builtin_amdgcn_tanh_f32(x);
#else
    return tanhf(x);
#endif
}

// This kernel runs exactly one 16-wave block per WGP (4 waves/SIMD).  Tell the
// compiler so: relaxes the register-pressure budget to file/4, which lets the
// scheduler keep a group's A-fragments live instead of serializing each
// ds_load pair against its consuming WMMA.
__global__ __launch_bounds__(NTHREADS)
__attribute__((amdgpu_waves_per_eu(4, 8)))
void rnn_fused_wmma(const float* __restrict__ x, const float* __restrict__ Wh,
                    const float* __restrict__ bh, const float* __restrict__ Wo,
                    const float* __restrict__ bo, float* __restrict__ out,
                    int Tlen) {
    __shared__ bf16 Ash[M_BLK * A_STRIDE];   // combined [x_t | h | pad] tile, bf16

    const int tid  = threadIdx.x;
    const int lane = tid & 31;
    const int wave = tid >> 5;
    const int g    = lane >> 4;        // lane half (K-interleave group)
    const int nloc = lane & 15;        // N within tile (B/C/D) and M row (A)
    const int n0   = wave * 16;        // this wave's N-tile base column
    const int b0   = blockIdx.x * M_BLK;

    // ---- init: zero h region + K pad (cols 300..575) -> h0 = 0 ----
    for (int i = tid; i < M_BLK * (K_PAD - D_INPUT); i += NTHREADS) {
        int m = i / (K_PAD - D_INPUT);
        int k = D_INPUT + i % (K_PAD - D_INPUT);
        Ash[m * A_STRIDE + k] = (bf16)0.0f;
    }

    // ---- preload all B fragments into registers (weights are step-invariant) ----
    // 16-bit B 32x16 layout: lane l holds n = l&15; element i holds
    // k = kt*32 + (i<8 ? 0 : 16) + (l>>4)*8 + (i&7).  Wh is [K_TOT][D_HID] row-major.
    v16bf Bfrag[KT];
#pragma unroll
    for (int kt = 0; kt < KT; ++kt) {
#pragma unroll
        for (int i = 0; i < 16; ++i) {
            int k = kt * 32 + ((i < 8) ? 0 : 16) + g * 8 + (i & 7);
            float w = (k < K_TOT) ? Wh[(size_t)k * D_HID + n0 + nloc] : 0.0f;
            Bfrag[kt][i] = (bf16)w;
        }
    }
    const float bias = bh[n0 + nloc];

    // A-fragment loader: 16-bit A 16x32 layout — lane l holds row m = l&15;
    // elements 0..7 -> k = kt*32 + g*8 .. +7, elements 8..15 -> +16.
    const bf16* abase = Ash + nloc * A_STRIDE + g * 8;
    auto loadA = [&](int kt) -> v16bf {
        const bf16* arow = abase + kt * 32;
        v8bf a_lo = *reinterpret_cast<const v8bf*>(arow);
        v8bf a_hi = *reinterpret_cast<const v8bf*>(arow + 16);
        v16bf a;
#pragma unroll
        for (int i = 0; i < 8; ++i) { a[i] = a_lo[i]; a[8 + i] = a_hi[i]; }
        return a;
    };

    // ---- x prefetch plumbing: 1200 float4 chunks per step, <=3 per thread ----
    const int c0 = tid, c1 = tid + NTHREADS, c2 = tid + 2 * NTHREADS;
    auto ldx = [&](int c, int t) -> float4 {
        int row = c / 75, col4 = c % 75;
        return *reinterpret_cast<const float4*>(
            x + ((size_t)(b0 + row) * (size_t)Tlen + (size_t)t) * D_INPUT + col4 * 4);
    };
    auto stx = [&](int c, float4 v) {
        int row = c / 75, col4 = c % 75;
        v4bf w;
        w[0] = (bf16)v.x; w[1] = (bf16)v.y; w[2] = (bf16)v.z; w[3] = (bf16)v.w;
        *reinterpret_cast<v4bf*>(Ash + row * A_STRIDE + col4 * 4) = w;
    };

    float4 xr0 = ldx(c0, 0);
    float4 xr1 = ldx(c1, 0);
    float4 xr2;
    if (c2 < NCHUNK) xr2 = ldx(c2, 0);

    // ---- recurrence ----
    for (int t = 0; t < Tlen; ++t) {
        // write x_t (bf16) into A tile cols 0..299
        stx(c0, xr0);
        stx(c1, xr1);
        if (c2 < NCHUNK) stx(c2, xr2);
        __syncthreads();                       // x_t + h_{t-1} visible to all waves

        // prefetch x_{t+1} while the matrix pipes work on step t
        if (t + 1 < Tlen) {
            xr0 = ldx(c0, t + 1);
            xr1 = ldx(c1, t + 1);
            if (c2 < NCHUNK) xr2 = ldx(c2, t + 1);
        }

        // [16 x 576] @ [576 x 16] for this wave's N tile.
        // Groups of 3 K-tiles: all three A fragments live across the first
        // WMMA -> distinct registers -> LDS latency of frags 1/2 hides under
        // WMMAs 0/1.
        v8f acc = {};
#pragma unroll
        for (int kg = 0; kg < KT; kg += KGRP) {
            v16bf a0 = loadA(kg + 0);
            v16bf a1 = loadA(kg + 1);
            v16bf a2 = loadA(kg + 2);
            acc = __builtin_amdgcn_wmma_f32_16x16x32_bf16(
                false, a0, false, Bfrag[kg + 0], (short)0, acc, false, false);
            acc = __builtin_amdgcn_wmma_f32_16x16x32_bf16(
                false, a1, false, Bfrag[kg + 1], (short)0, acc, false, false);
            acc = __builtin_amdgcn_wmma_f32_16x16x32_bf16(
                false, a2, false, Bfrag[kg + 2], (short)0, acc, false, false);
        }
        __syncthreads();                       // all A reads done before h_t writes

        // h_t = tanh(clip(pre, +-4)); C/D layout: lane l holds n=l&15, rows r+8*(l>>4)
#pragma unroll
        for (int r = 0; r < 8; ++r) {
            int m = r + 8 * g;
            float pre = acc[r] + bias;
            pre = fminf(fmaxf(pre, -4.0f), 4.0f);
            float hv = fast_tanh(pre);
            Ash[m * A_STRIDE + D_INPUT + n0 + nloc] = (bf16)hv;
        }
    }
    __syncthreads();

    // ---- head: h_final @ Wo + bo, clip, log_softmax (D_OUT = 2) ----
    if (tid < M_BLK) {
        const int m = tid;
        float p0 = bo[0], p1 = bo[1];
        for (int j = 0; j < D_HID; ++j) {
            float hv = (float)Ash[m * A_STRIDE + D_INPUT + j];
            p0 = fmaf(hv, Wo[j * 2 + 0], p0);
            p1 = fmaf(hv, Wo[j * 2 + 1], p1);
        }
        p0 = fminf(fmaxf(p0, -4.0f), 4.0f);
        p1 = fminf(fmaxf(p1, -4.0f), 4.0f);
        float mx  = fmaxf(p0, p1);
        float lse = mx + logf(expf(p0 - mx) + expf(p1 - mx));
        out[(size_t)(b0 + m) * 2 + 0] = p0 - lse;
        out[(size_t)(b0 + m) * 2 + 1] = p1 - lse;
    }
}

extern "C" void kernel_launch(void* const* d_in, const int* in_sizes, int n_in,
                              void* d_out, int out_size, void* d_ws, size_t ws_size,
                              hipStream_t stream) {
    const float* x  = (const float*)d_in[0];   // [1024, 200, 300] f32
    const float* Wh = (const float*)d_in[1];   // [556, 256] f32
    const float* bh = (const float*)d_in[2];   // [256] f32
    const float* Wo = (const float*)d_in[3];   // [256, 2] f32
    const float* bo = (const float*)d_in[4];   // [2] f32
    float* out = (float*)d_out;                // [1024, 2] f32

    dim3 grid(1024 / M_BLK);                   // 64 blocks, each owns 16 batch rows
    rnn_fused_wmma<<<grid, NTHREADS, 0, stream>>>(x, Wh, bh, Wo, bo, out, T_STEPS);
}
